// GraphAttentionNodeFlow_23519240912983
// MI455X (gfx1250) — matrix operand order repaired and checked by reading, not compiled, not run on hardware
//
#include <hip/hip_runtime.h>
#include <hip/hip_bf16.h>

typedef __attribute__((ext_vector_type(16))) _Float16 v16h;
typedef __attribute__((ext_vector_type(8)))  float    v8f;

#define N_SRC   100000
#define N_DST   25000
#define E_NUM   800000
#define IN_DIM  256
#define H_NUM   8
#define D_DIM   32
#define OUT_DIM 256           // H_NUM * D_DIM
#define ALPHA   0.2f

// f16 copy of W, grouped for immediate-offset DS addressing:
//   row o of W lives at halves[(o%16)*WH_GROUP + (o/16)*WH_SLOT]
//   WH_SLOT  = 264 halves (528B: 256 data + pad, 16B multiple)
//   WH_GROUP = 16*264 + 8 = 4232 halves (8464B: 16 slots + 16B pad ->
//              lane dword stride 2116 = 4 mod 64 -> 16 distinct banks)
#define WH_SLOT   264
#define WH_GROUP  4232
#define WH_HALVES (16 * WH_GROUP)               // 67,712 halves = 135,424 B

// ---------------------------------------------------------------------------
// Workspace layout (bytes, all 256B aligned):
//   ft   : [N_SRC, 256] f32   off 0            size 102,400,000
//   a1   : [N_SRC, 8]   f32   off 102,400,000  size   3,200,000
//   a2   : [N_DST, 8]   f32   off 105,600,000  size     800,000
//   norm : [N_DST, 8]   f32   off 106,400,000  size     800,000
//   agg  : [N_DST, 256] f32   off 107,200,000  size  25,600,000
//   Wh   : grouped f16        off 132,800,000  size     135,424
// ---------------------------------------------------------------------------
#define OFF_FT   0
#define OFF_A1   102400000
#define OFF_A2   105600000
#define OFF_NORM 106400000
#define OFF_AGG  107200000
#define OFF_WH   132800000

// Zero normalizer + agg (contiguous region: 6,600,000 floats) every call.
__global__ __launch_bounds__(256) void gat_zero_kernel(float4* p, int n4) {
    int t = blockIdx.x * blockDim.x + threadIdx.x;
    if (t < n4) p[t] = make_float4(0.f, 0.f, 0.f, 0.f);
}

// W f32 -> grouped f16 blob (layout documented above). One thread/element.
__global__ __launch_bounds__(256) void gat_wcvt_kernel(
        const float* __restrict__ W, _Float16* __restrict__ Wh) {
    const int t = blockIdx.x * blockDim.x + threadIdx.x;   // 0..65535
    const int o = t >> 8;
    const int k = t & 255;
    Wh[(o & 15) * WH_GROUP + (o >> 4) * WH_SLOT + k] = (_Float16)W[t];
}

// A-matrix fragment (16x32 f16): lanes 0-15 hold K = kb+0..15 of
// row M=lane; lanes 16-31 hold K = kb+16..31 of row M=lane-16.
__device__ __forceinline__ v16h load_a_frag(const float* __restrict__ row,
                                            int g0, int g1) {
    const float4 x0 = *reinterpret_cast<const float4*>(row + g0);
    const float4 x1 = *reinterpret_cast<const float4*>(row + g0 + 4);
    const float4 y0 = *reinterpret_cast<const float4*>(row + g1);
    const float4 y1 = *reinterpret_cast<const float4*>(row + g1 + 4);
    v16h a;
    a[0]  = (_Float16)x0.x; a[1]  = (_Float16)x0.y;
    a[2]  = (_Float16)x0.z; a[3]  = (_Float16)x0.w;
    a[4]  = (_Float16)x1.x; a[5]  = (_Float16)x1.y;
    a[6]  = (_Float16)x1.z; a[7]  = (_Float16)x1.w;
    a[8]  = (_Float16)y0.x; a[9]  = (_Float16)y0.y;
    a[10] = (_Float16)y0.z; a[11] = (_Float16)y0.w;
    a[12] = (_Float16)y1.x; a[13] = (_Float16)y1.y;
    a[14] = (_Float16)y1.z; a[15] = (_Float16)y1.w;
    return a;
}

// B fragment from LDS: 32B contiguous, 16B aligned -> 2x ds_load_b128.
// Byte offset from the lane base = n*528 + 2*kb <= 8368: always a 16-bit DS
// immediate, so the WMMA chain contains no address VALU (and no WAR v_nops).
__device__ __forceinline__ v16h load_b_frag(const _Float16* bbase,
                                            int n, int kb) {
    return *reinterpret_cast<const v16h*>(bbase + n * WH_SLOT + kb);
}

// ft = h @ W^T.  One wave32 owns a 16x256 output row-slab: loads each A
// fragment once and runs 16 WMMAs against it (16 v8f accumulators).  W lives
// in LDS as f16, staged with global_load_async_to_lds_b128 (ASYNCcnt path).
// B fragments are pipelined at distance 2 (3 rotating octets): ds_loads for
// tile n+2 issue before the WMMA of tile n, so waits are partial
// (s_wait_dscnt 0x4) and the ds write target is outside the hazard window.
__global__ __launch_bounds__(256) void gat_gemm_wmma_kernel(
        const float* __restrict__ h, const _Float16* __restrict__ Wh,
        float* __restrict__ ft) {
    __shared__ __align__(16) _Float16 lds_w[WH_HALVES];

    // ---- async stage Wh -> LDS (135,424 B = 8464 x 16B chunks) ----
    {
        const uint32_t lbase = (uint32_t)(uintptr_t)(&lds_w[0]);
        const uint64_t gbase = (uint64_t)(uintptr_t)Wh;
        for (int i = threadIdx.x; i < (WH_HALVES * 2) / 16; i += 256) {
            const uint32_t l = lbase + (uint32_t)i * 16u;
            const uint64_t g = gbase + (uint64_t)i * 16u;
            asm volatile("global_load_async_to_lds_b128 %0, %1, off"
                         :: "v"(l), "v"(g) : "memory");
        }
        asm volatile("s_wait_asynccnt 0x0" ::: "memory");
    }
    __syncthreads();

    const int lane = threadIdx.x & 31;
    const int wave = threadIdx.x >> 5;
    const int idx  = lane & 15;
    const int hsel = lane >> 4;
    const int mTile = blockIdx.x * 8 + wave;           // 0 .. 6249 (+guard)
    if (mTile >= N_SRC / 16) return;

    const float* __restrict__ Arow = h + (size_t)(mTile * 16 + idx) * IN_DIM;

    // Lane base: group idx (W rows o with o%16==idx), K half-select.
    const _Float16* bbase = &lds_w[idx * WH_GROUP + hsel * 16];

    v8f acc[16];
#pragma unroll
    for (int n = 0; n < 16; ++n)
        acc[n] = (v8f){0.f, 0.f, 0.f, 0.f, 0.f, 0.f, 0.f, 0.f};

#pragma unroll
    for (int kb = 0; kb < IN_DIM; kb += 32) {
        if (kb + 32 < IN_DIM) {
            __builtin_prefetch(Arow + kb + 32, 0, 1);  // global_prefetch_b8
        }
        const int g0 = kb + hsel * 16;                 // A: K 0..15 | 16..31
        const v16h a = load_a_frag(Arow, g0, g0 + 8);

        v16h bq0 = load_b_frag(bbase, 0, kb);
        v16h bq1 = load_b_frag(bbase, 1, kb);
#pragma unroll
        for (int n = 0; n < 16; ++n) {
            v16h bq2 = bq0;
            if (n < 14) bq2 = load_b_frag(bbase, n + 2, kb);
            acc[n] = __builtin_amdgcn_wmma_f32_16x16x32_f16(
                         /*neg_a=*/false, a, /*neg_b=*/false, bq0,
                         /*c_mod=*/(short)0, acc[n],
                         /*reuse_a=*/false, /*reuse_b=*/false);
            bq0 = bq1;
            bq1 = bq2;
        }
    }

    // C/D layout: VGPR r -> (M = r + 8*hsel, N = idx)
    float* __restrict__ outBase = ft + (size_t)mTile * 16 * OUT_DIM;
#pragma unroll
    for (int n = 0; n < 16; ++n) {
#pragma unroll
        for (int r = 0; r < 8; ++r) {
            outBase[(size_t)(r + hsel * 8) * OUT_DIM + n * 16 + idx] = acc[n][r];
        }
    }
}

// a1[n,h] = <ft[n,h,:], attn_l[h,:]>;  a2 (n < N_DST) with attn_r.
__global__ __launch_bounds__(256) void gat_logits_kernel(
        const float* __restrict__ ft,
        const float* __restrict__ attn_l, const float* __restrict__ attn_r,
        float* __restrict__ a1, float* __restrict__ a2) {
    const int t = blockIdx.x * blockDim.x + threadIdx.x;   // (node, head)
    if (t >= N_SRC * H_NUM) return;
    const int n  = t >> 3;
    const int hh = t & 7;
    const float4* f  = reinterpret_cast<const float4*>(ft + (size_t)n * OUT_DIM + hh * D_DIM);
    const float4* al = reinterpret_cast<const float4*>(attn_l + hh * D_DIM);
    const float4* ar = reinterpret_cast<const float4*>(attn_r + hh * D_DIM);
    float s1 = 0.f, s2 = 0.f;
#pragma unroll
    for (int i = 0; i < D_DIM / 4; ++i) {
        const float4 v = f[i], l = al[i], r = ar[i];
        s1 += v.x * l.x + v.y * l.y + v.z * l.z + v.w * l.w;
        s2 += v.x * r.x + v.y * r.y + v.z * r.z + v.w * r.w;
    }
    a1[t] = s1;
    if (n < N_DST) a2[n * H_NUM + hh] = s2;
}

// One wave32 per edge; lane = feature dim d.  ft (102MB) + agg (25.6MB) live
// in the 192MB L2, so gathers and f32 atomics resolve at L2 bandwidth.
__global__ __launch_bounds__(256) void gat_edge_kernel(
        const int* __restrict__ src, const int* __restrict__ dst,
        const float* __restrict__ ft,
        const float* __restrict__ a1, const float* __restrict__ a2,
        float* __restrict__ norm, float* __restrict__ agg) {
    const int lane = threadIdx.x & 31;
    const int e = blockIdx.x * 8 + (threadIdx.x >> 5);
    if (e >= E_NUM) return;
    const int s = src[e];
    const int d = dst[e];
    const float* __restrict__ fs = ft  + (size_t)s * OUT_DIM;
    float*       __restrict__ ag = agg + (size_t)d * OUT_DIM;
#pragma unroll
    for (int hh = 0; hh < H_NUM; ++hh) {
        const float logit = a1[s * H_NUM + hh] + a2[d * H_NUM + hh];
        const float lrelu = logit > 0.f ? logit : ALPHA * logit;
        const float sc    = __expf(lrelu);                 // v_exp_f32
        const float v     = fs[hh * D_DIM + lane];         // coalesced 128B
        atomicAdd(&ag[hh * D_DIM + lane], v * sc);         // global_atomic_add_f32
        if (lane == 0) atomicAdd(&norm[d * H_NUM + hh], sc);
    }
}

// ret = agg / normalizer + ret_bias
__global__ __launch_bounds__(256) void gat_finalize_kernel(
        const float* __restrict__ agg, const float* __restrict__ norm,
        const float* __restrict__ ret_bias, float* __restrict__ out) {
    const int t = blockIdx.x * blockDim.x + threadIdx.x;
    if (t >= N_DST * OUT_DIM) return;
    const int n  = t >> 8;
    const int c  = t & 255;
    const int hh = c >> 5;
    out[t] = agg[t] / norm[n * H_NUM + hh] + ret_bias[c];
}

extern "C" void kernel_launch(void* const* d_in, const int* in_sizes, int n_in,
                              void* d_out, int out_size, void* d_ws, size_t ws_size,
                              hipStream_t stream) {
    (void)in_sizes; (void)n_in; (void)out_size; (void)ws_size;
    const float* h       = (const float*)d_in[0];
    const int*   src     = (const int*)  d_in[1];
    const int*   dst     = (const int*)  d_in[2];
    const float* W       = (const float*)d_in[3];
    const float* attn_l  = (const float*)d_in[4];
    const float* attn_r  = (const float*)d_in[5];
    const float* ret_bias= (const float*)d_in[6];
    float* out = (float*)d_out;

    char* ws = (char*)d_ws;
    float*     ft   = (float*)    (ws + OFF_FT);
    float*     a1   = (float*)    (ws + OFF_A1);
    float*     a2   = (float*)    (ws + OFF_A2);
    float*     norm = (float*)    (ws + OFF_NORM);
    float*     agg  = (float*)    (ws + OFF_AGG);
    _Float16*  Wh   = (_Float16*) (ws + OFF_WH);

    // 1) zero the atomic-scatter buffers (norm + agg are contiguous)
    {
        const int n4 = (N_DST * H_NUM + N_DST * OUT_DIM) / 4;   // 1,650,000
        gat_zero_kernel<<<(n4 + 255) / 256, 256, 0, stream>>>((float4*)norm, n4);
    }
    // 2) W f32 -> grouped f16 blob (LDS-stageable)
    gat_wcvt_kernel<<<(256 * 256) / 256, 256, 0, stream>>>(W, Wh);
    // 3) WMMA GEMM: ft = h @ W^T  (6250 M-tiles, 8 waves/block)
    gat_gemm_wmma_kernel<<<(N_SRC / 16 + 7) / 8, 256, 0, stream>>>(h, Wh, ft);
    // 4) per-(node, head) attention logits
    gat_logits_kernel<<<(N_SRC * H_NUM + 255) / 256, 256, 0, stream>>>(
        ft, attn_l, attn_r, a1, a2);
    // 5) edge gather-multiply-scatter (one wave per edge)
    gat_edge_kernel<<<E_NUM / 8, 256, 0, stream>>>(
        src, dst, ft, a1, a2, norm, agg);
    // 6) normalize + bias
    gat_finalize_kernel<<<(N_DST * OUT_DIM + 255) / 256, 256, 0, stream>>>(
        agg, norm, ret_bias, out);
}
